// ExtremePooling_82016695485108
// MI455X (gfx1250) — matrix operand order repaired and checked by reading, not compiled, not run on hardware
//
#include <hip/hip_runtime.h>
#include <hip/hip_bf16.h>
#include <math.h>
#include <stdint.h>

// ---------------------------------------------------------------------------
// CDNA5 (gfx1250): implicit-GEMM convs via v_wmma_f32_16x16x32_f16 with
// double-buffered LDS fed by GLOBAL_LOAD_ASYNC_TO_LDS_B128 (ASYNCcnt).
// ---------------------------------------------------------------------------

typedef __attribute__((ext_vector_type(16))) _Float16 v16h;
typedef __attribute__((ext_vector_type(8)))  _Float16 v8h;
typedef __attribute__((ext_vector_type(8)))  float    v8f;

union AFrag { v16h v; v8h h[2]; };

#define BATCH 8
#define CIN   256
#define CMID  128
#define COUT  256
#define HW    128
#define HP    130     // padded height  (h' = h+1)
#define WP    136     // padded width   (w' = w+1; extra cols for async divisibility)

// 16 bytes per lane, global -> LDS, tracked by ASYNCcnt
__device__ __forceinline__ void async_b128(uint32_t lds_off, uint64_t sbase, uint32_t voff) {
    asm volatile("global_load_async_to_lds_b128 %0, %1, %2"
                 :: "v"(lds_off), "v"(voff), "s"(sbase) : "memory");
}
#define WAIT_ASYNC_16() asm volatile("s_wait_asynccnt 16" ::: "memory")
#define WAIT_ASYNC_3()  asm volatile("s_wait_asynccnt 3"  ::: "memory")
#define WAIT_ASYNC_0()  asm volatile("s_wait_asynccnt 0"  ::: "memory")

#define WMMA(A, B, C) __builtin_amdgcn_wmma_f32_16x16x32_f16(false, (A), false, (B), (short)0, (C), false, false)

// ===========================================================================
// pack_x: NCHW f32 -> zero-padded NHWC f16  xp[n][130][136][256]
// grid = (130, 8), 256 threads
// ===========================================================================
__global__ __launch_bounds__(256)
void pack_x_kernel(const float* __restrict__ x, _Float16* __restrict__ xp)
{
    const int hp = blockIdx.x, n = blockIdx.y, t = threadIdx.x;
    _Float16* dst = xp + (((size_t)n * HP + hp) * WP) * CIN;
    const int hi = hp - 1;
    for (int e = t; e < WP * CIN; e += 256) {
        int c = e / WP, wp = e % WP;
        int wi = wp - 1;
        float v = 0.0f;
        if (hi >= 0 && hi < HW && wi >= 0 && wi < HW)
            v = x[(((size_t)n * CIN + c) * HW + hi) * HW + wi];
        dst[(size_t)wp * CIN + c] = (_Float16)v;
    }
}

// ===========================================================================
// pack_w: all weights -> f16 in LDS-tile order, BN scale folded in.
//   wp1   [ct8][cc8][m64][tap9][k32]  (four branch convs, concat 512 out ch)
//   wp2   [ct4][cc4][m64][tap9][k32]  (*s2)
//   wp1x1 [ct4][cc8][m64][k32]        (*s1)
// ===========================================================================
#define NW1 (8 * 8 * 64 * 9 * 32)
#define NW2 (4 * 4 * 64 * 9 * 32)
#define NW3 (4 * 8 * 64 * 32)
__global__ __launch_bounds__(256)
void pack_w_kernel(const float* __restrict__ w_l, const float* __restrict__ s_l,
                   const float* __restrict__ w_r, const float* __restrict__ s_r,
                   const float* __restrict__ w_t, const float* __restrict__ s_t,
                   const float* __restrict__ w_b, const float* __restrict__ s_b,
                   const float* __restrict__ w2,  const float* __restrict__ s2,
                   const float* __restrict__ w1,  const float* __restrict__ s1,
                   _Float16* __restrict__ wp1, _Float16* __restrict__ wp2,
                   _Float16* __restrict__ wp1x1)
{
    int idx = blockIdx.x * 256 + threadIdx.x;
    if (idx < NW1) {
        int k = idx % 32, tap = (idx / 32) % 9, m = (idx / 288) % 64;
        int cc = (idx / (288 * 64)) % 8, ct = idx / (288 * 64 * 8);
        int branch = ct >> 1, co = (ct & 1) * 64 + m, ci = cc * 32 + k;
        const float* w = (branch == 0) ? w_l : (branch == 1) ? w_r : (branch == 2) ? w_t : w_b;
        const float* s = (branch == 0) ? s_l : (branch == 1) ? s_r : (branch == 2) ? s_t : s_b;
        wp1[idx] = (_Float16)(w[((size_t)co * CIN + ci) * 9 + tap] * s[co]);
    } else if (idx < NW1 + NW2) {
        int j = idx - NW1;
        int k = j % 32, tap = (j / 32) % 9, m = (j / 288) % 64;
        int cc = (j / (288 * 64)) % 4, ct = j / (288 * 64 * 4);
        int co = ct * 64 + m, ci = cc * 32 + k;
        wp2[j] = (_Float16)(w2[((size_t)co * CMID + ci) * 9 + tap] * s2[co]);
    } else if (idx < NW1 + NW2 + NW3) {
        int j = idx - NW1 - NW2;
        int k = j % 32, m = (j / 32) % 64, cc = (j / 2048) % 8, ct = j / (2048 * 8);
        int co = ct * 64 + m, ci = cc * 32 + k;
        wp1x1[j] = (_Float16)(w1[(size_t)co * CIN + ci] * s1[co]);
    }
}

// ===========================================================================
// Phase 1: four 3x3 convs (256 -> 512 out ch) + BN + ReLU, implicit GEMM.
// Block tile M=64 x N=128 (one row). 256 thr = 8 waves (2M x 4N), wave tile
// 32x32 (2x2 WMMA frags). Double-buffered LDS via async copies.
// Output y[n][branch][ch][h][w] f16.
// ===========================================================================
__global__ __launch_bounds__(256)
void branch_conv_kernel(const _Float16* __restrict__ xp,
                        const _Float16* __restrict__ wp1,
                        const float* __restrict__ o_l, const float* __restrict__ o_r,
                        const float* __restrict__ o_t, const float* __restrict__ o_b,
                        _Float16* __restrict__ y)
{
    __shared__ __attribute__((aligned(64))) _Float16 Bs[2][3][WP][32];   // 52,224 B
    __shared__ __attribute__((aligned(64))) _Float16 As[2][64][9][32];   // 73,728 B

    const int ct = blockIdx.x;            // 0..7 -> 64-ch tile of 512
    const int h  = blockIdx.y;
    const int n  = blockIdx.z;
    const int t  = threadIdx.x;
    const int lane  = t & 31;
    const int wid   = t >> 5;
    const int waveM = wid & 1;            // 2 waves x 32 rows
    const int waveN = wid >> 1;           // 4 waves x 32 cols

    const int branch  = ct >> 1;
    const int co_base = (ct & 1) * 64;
    const float* osel = (branch == 0) ? o_l : (branch == 1) ? o_r : (branch == 2) ? o_t : o_b;

    const uint32_t bs_base = (uint32_t)(uintptr_t)&Bs[0][0][0][0];
    const uint32_t as_base = (uint32_t)(uintptr_t)&As[0][0][0][0];

    const int mrow = lane & 15;
    const int kbA  = (lane & 16) ? 16 : 0;
    const int kbB  = (lane & 16) ? 16 : 0;
    const int ncol = lane & 15;
    const int laneq = lane & 3;           // 16B quarter of a 64B cell

    v8f acc[2][2];
    for (int i = 0; i < 2; ++i) for (int j = 0; j < 2; ++j)
        for (int r = 0; r < 8; ++r) acc[i][j][r] = 0.0f;

    const uint64_t ximg = (uint64_t)(uintptr_t)xp
                        + ((size_t)n * HP + h) * WP * CIN * sizeof(_Float16);
    const uint64_t wbase = (uint64_t)(uintptr_t)wp1 + (size_t)ct * 8 * 36864;

    // async prefetch of K-stage cc into buffer cc&1: 16 issues per wave
    auto prefetch = [&](int cc) {
        const int buf = cc & 1;
        const uint64_t srcB = ximg + (size_t)cc * 32 * sizeof(_Float16);
        for (int j = 0; j < 7; ++j) {                     // B slab: 51 instrs, 7/wave
            int cell = (wid * 7 + j) * 8 + (lane >> 2);   // (r,c) over [3][136]
            if (cell > 3 * WP - 1) cell = 3 * WP - 1;
            uint32_t voff = (uint32_t)cell * (CIN * 2) + laneq * 16;
            uint32_t ldst = bs_base + (uint32_t)(buf * 3 * WP + cell) * 64 + laneq * 16;
            async_b128(ldst, srcB, voff);
        }
        const uint64_t srcA = wbase + (size_t)cc * 36864;
        for (int j = 0; j < 9; ++j) {                     // A tile: 72 instrs, 9/wave
            uint32_t off = (uint32_t)((wid * 9 + j) * 512 + lane * 16);
            async_b128(as_base + buf * 36864 + off, srcA, off);
        }
    };

    auto compute = [&](int buf) {
#pragma unroll
        for (int tap = 0; tap < 9; ++tap) {
            const int kh = tap / 3, kw = tap % 3;
            AFrag a0, a1;
            a0.h[0] = *(const v8h*)&As[buf][waveM * 32 + mrow][tap][kbA];
            a0.h[1] = *(const v8h*)&As[buf][waveM * 32 + mrow][tap][kbA + 8];
            a1.h[0] = *(const v8h*)&As[buf][waveM * 32 + 16 + mrow][tap][kbA];
            a1.h[1] = *(const v8h*)&As[buf][waveM * 32 + 16 + mrow][tap][kbA + 8];
            v16h b0 = *(const v16h*)&Bs[buf][kh][waveN * 32 + ncol + kw][kbB];
            v16h b1 = *(const v16h*)&Bs[buf][kh][waveN * 32 + 16 + ncol + kw][kbB];
            acc[0][0] = WMMA(a0.v, b0, acc[0][0]);
            acc[0][1] = WMMA(a0.v, b1, acc[0][1]);
            acc[1][0] = WMMA(a1.v, b0, acc[1][0]);
            acc[1][1] = WMMA(a1.v, b1, acc[1][1]);
        }
    };

    prefetch(0);
    for (int cc = 0; cc < CIN / 32; ++cc) {
        if (cc + 1 < CIN / 32) { prefetch(cc + 1); WAIT_ASYNC_16(); }
        else                   { WAIT_ASYNC_0(); }
        __syncthreads();
        compute(cc & 1);
        __syncthreads();
    }

    const int mofs = (lane & 16) ? 8 : 0;
#pragma unroll
    for (int mf = 0; mf < 2; ++mf)
#pragma unroll
        for (int nf = 0; nf < 2; ++nf) {
            const int pix = waveN * 32 + nf * 16 + (lane & 15);
#pragma unroll
            for (int r = 0; r < 8; ++r) {
                int co = co_base + waveM * 32 + mf * 16 + r + mofs;   // 0..127 in branch
                float v = fmaxf(acc[mf][nf][r] + osel[co], 0.0f);
                y[((((size_t)n * 4 + branch) * CMID + co) * HW + h) * HW + pix] = (_Float16)v;
            }
        }
}

// ===========================================================================
// Phase 2: corner pooling; writes zero-padded NHWC f16 sp[n][130][136][128].
// grid = (128 channels, 8 images), 128 threads, 64 KB LDS f32 tile.
// ===========================================================================
__global__ __launch_bounds__(128)
void corner_pool_kernel(const _Float16* __restrict__ y, _Float16* __restrict__ sp)
{
    __shared__ float tile[HW * HW];

    const int ch = blockIdx.x;
    const int n  = blockIdx.y;
    const int t  = threadIdx.x;

    const size_t plane = (size_t)HW * HW;
    const _Float16* yl = y + ((size_t)(n * 4 + 0) * CMID + ch) * plane;
    const _Float16* yr = y + ((size_t)(n * 4 + 1) * CMID + ch) * plane;
    const _Float16* yt = y + ((size_t)(n * 4 + 2) * CMID + ch) * plane;
    const _Float16* yb = y + ((size_t)(n * 4 + 3) * CMID + ch) * plane;

    {   // row scans (thread = row)
        const _Float16* rl = yl + (size_t)t * HW;
        const _Float16* rr = yr + (size_t)t * HW;
        float run = -3.4e38f;
        for (int w = 0; w < HW; ++w)  { run = fmaxf(run, (float)rr[w]); tile[t * HW + w]  = run; }
        run = -3.4e38f;
        for (int w = HW - 1; w >= 0; --w) { run = fmaxf(run, (float)rl[w]); tile[t * HW + w] += run; }
    }
    __syncthreads();
    {   // column scans (thread = col)
        float run = -3.4e38f;
        for (int hh = 0; hh < HW; ++hh)     { run = fmaxf(run, (float)yb[(size_t)hh * HW + t]); tile[hh * HW + t] += run; }
        run = -3.4e38f;
        for (int hh = HW - 1; hh >= 0; --hh){ run = fmaxf(run, (float)yt[(size_t)hh * HW + t]); tile[hh * HW + t] += run; }
    }
    __syncthreads();

    _Float16* dst = sp + (size_t)n * HP * WP * CMID;
    for (int e = t; e < HP * WP; e += 128) {
        int hp = e / WP, wp = e % WP;
        float v = 0.0f;
        if (hp >= 1 && hp <= HW && wp >= 1 && wp <= HW)
            v = tile[(hp - 1) * HW + (wp - 1)];
        dst[(size_t)e * CMID + ch] = (_Float16)v;
    }
}

// ===========================================================================
// Phase 3: out = relu( conv3x3(sp, w2*s2) + conv1x1(xp, w1*s1) + o2 + o1 )
// Same tiling; both GEMMs accumulate into one register tile.
// grid = (4, 128, 8), 256 threads.
// ===========================================================================
__global__ __launch_bounds__(256)
void fuse_conv_kernel(const _Float16* __restrict__ sp,
                      const _Float16* __restrict__ xp,
                      const _Float16* __restrict__ wp2,
                      const _Float16* __restrict__ wp1x1,
                      const float* __restrict__ o2, const float* __restrict__ o1,
                      float* __restrict__ out)
{
    __shared__ __attribute__((aligned(64))) _Float16 Bs[2][3][WP][32];   // 52,224 B
    __shared__ __attribute__((aligned(64))) _Float16 As[2][64][9][32];   // 73,728 B
    __shared__ __attribute__((aligned(64))) _Float16 B1[2][HW][32];      // 16,384 B
    __shared__ __attribute__((aligned(64))) _Float16 A1[2][64][32];      //  8,192 B

    const int ct = blockIdx.x;
    const int h  = blockIdx.y;
    const int n  = blockIdx.z;
    const int t  = threadIdx.x;
    const int lane  = t & 31;
    const int wid   = t >> 5;
    const int waveM = wid & 1;
    const int waveN = wid >> 1;

    const uint32_t bs_base = (uint32_t)(uintptr_t)&Bs[0][0][0][0];
    const uint32_t as_base = (uint32_t)(uintptr_t)&As[0][0][0][0];
    const uint32_t b1_base = (uint32_t)(uintptr_t)&B1[0][0][0];
    const uint32_t a1_base = (uint32_t)(uintptr_t)&A1[0][0][0];

    const int mrow = lane & 15;
    const int kbA  = (lane & 16) ? 16 : 0;
    const int kbB  = (lane & 16) ? 16 : 0;
    const int ncol = lane & 15;
    const int laneq = lane & 3;

    v8f acc[2][2];
    for (int i = 0; i < 2; ++i) for (int j = 0; j < 2; ++j)
        for (int r = 0; r < 8; ++r) acc[i][j][r] = 0.0f;

    const uint64_t simg = (uint64_t)(uintptr_t)sp
                        + ((size_t)n * HP + h) * WP * CMID * sizeof(_Float16);
    const uint64_t w2base = (uint64_t)(uintptr_t)wp2 + (size_t)ct * 4 * 36864;
    const uint64_t ximg = (uint64_t)(uintptr_t)xp
                        + (((size_t)n * HP + h + 1) * WP + 1) * CIN * sizeof(_Float16);
    const uint64_t w1base = (uint64_t)(uintptr_t)wp1x1 + (size_t)ct * 8 * 4096;

    auto prefetch2 = [&](int cc) {           // 3x3 over sp: 16 issues per wave
        const int buf = cc & 1;
        const uint64_t srcB = simg + (size_t)cc * 32 * sizeof(_Float16);
        for (int j = 0; j < 7; ++j) {
            int cell = (wid * 7 + j) * 8 + (lane >> 2);
            if (cell > 3 * WP - 1) cell = 3 * WP - 1;
            uint32_t voff = (uint32_t)cell * (CMID * 2) + laneq * 16;
            uint32_t ldst = bs_base + (uint32_t)(buf * 3 * WP + cell) * 64 + laneq * 16;
            async_b128(ldst, srcB, voff);
        }
        const uint64_t srcA = w2base + (size_t)cc * 36864;
        for (int j = 0; j < 9; ++j) {
            uint32_t off = (uint32_t)((wid * 9 + j) * 512 + lane * 16);
            async_b128(as_base + buf * 36864 + off, srcA, off);
        }
    };
    auto prefetch1 = [&](int cc) {           // 1x1 over xp: 3 issues per wave
        const int buf = cc & 1;
        const uint64_t srcB = ximg + (size_t)cc * 32 * sizeof(_Float16);
        for (int j = 0; j < 2; ++j) {        // 16 instrs total
            int cell = (wid * 2 + j) * 8 + (lane >> 2);   // 0..127
            uint32_t voff = (uint32_t)cell * (CIN * 2) + laneq * 16;
            uint32_t ldst = b1_base + (uint32_t)(buf * HW + cell) * 64 + laneq * 16;
            async_b128(ldst, srcB, voff);
        }
        uint32_t off = (uint32_t)(wid * 512 + lane * 16); // 8 instrs total
        async_b128(a1_base + buf * 4096 + off, w1base + (size_t)cc * 4096 + off, off);
    };

    auto compute2 = [&](int buf) {
#pragma unroll
        for (int tap = 0; tap < 9; ++tap) {
            const int kh = tap / 3, kw = tap % 3;
            AFrag a0, a1;
            a0.h[0] = *(const v8h*)&As[buf][waveM * 32 + mrow][tap][kbA];
            a0.h[1] = *(const v8h*)&As[buf][waveM * 32 + mrow][tap][kbA + 8];
            a1.h[0] = *(const v8h*)&As[buf][waveM * 32 + 16 + mrow][tap][kbA];
            a1.h[1] = *(const v8h*)&As[buf][waveM * 32 + 16 + mrow][tap][kbA + 8];
            v16h b0 = *(const v16h*)&Bs[buf][kh][waveN * 32 + ncol + kw][kbB];
            v16h b1 = *(const v16h*)&Bs[buf][kh][waveN * 32 + 16 + ncol + kw][kbB];
            acc[0][0] = WMMA(a0.v, b0, acc[0][0]);
            acc[0][1] = WMMA(a0.v, b1, acc[0][1]);
            acc[1][0] = WMMA(a1.v, b0, acc[1][0]);
            acc[1][1] = WMMA(a1.v, b1, acc[1][1]);
        }
    };
    auto compute1 = [&](int buf) {
        AFrag a0, a1;
        a0.h[0] = *(const v8h*)&A1[buf][waveM * 32 + mrow][kbA];
        a0.h[1] = *(const v8h*)&A1[buf][waveM * 32 + mrow][kbA + 8];
        a1.h[0] = *(const v8h*)&A1[buf][waveM * 32 + 16 + mrow][kbA];
        a1.h[1] = *(const v8h*)&A1[buf][waveM * 32 + 16 + mrow][kbA + 8];
        v16h b0 = *(const v16h*)&B1[buf][waveN * 32 + ncol][kbB];
        v16h b1 = *(const v16h*)&B1[buf][waveN * 32 + 16 + ncol][kbB];
        acc[0][0] = WMMA(a0.v, b0, acc[0][0]);
        acc[0][1] = WMMA(a0.v, b1, acc[0][1]);
        acc[1][0] = WMMA(a1.v, b0, acc[1][0]);
        acc[1][1] = WMMA(a1.v, b1, acc[1][1]);
    };

    prefetch2(0);
    for (int cc = 0; cc < CMID / 32; ++cc) {
        if (cc + 1 < CMID / 32) { prefetch2(cc + 1); WAIT_ASYNC_16(); }
        else                    { prefetch1(0);      WAIT_ASYNC_3();  }
        __syncthreads();
        compute2(cc & 1);
        __syncthreads();
    }
    for (int cc = 0; cc < CIN / 32; ++cc) {
        if (cc + 1 < CIN / 32) { prefetch1(cc + 1); WAIT_ASYNC_3(); }
        else                   { WAIT_ASYNC_0(); }
        __syncthreads();
        compute1(cc & 1);
        __syncthreads();
    }

    const int mofs = (lane & 16) ? 8 : 0;
#pragma unroll
    for (int mf = 0; mf < 2; ++mf)
#pragma unroll
        for (int nf = 0; nf < 2; ++nf) {
            const int pix = waveN * 32 + nf * 16 + (lane & 15);
#pragma unroll
            for (int r = 0; r < 8; ++r) {
                int co = ct * 64 + waveM * 32 + mf * 16 + r + mofs;
                float v = fmaxf(acc[mf][nf][r] + o2[co] + o1[co], 0.0f);
                out[(((size_t)n * COUT + co) * HW + h) * HW + pix] = v;
            }
        }
}

// ===========================================================================
extern "C" void kernel_launch(void* const* d_in, const int* in_sizes, int n_in,
                              void* d_out, int out_size, void* d_ws, size_t ws_size,
                              hipStream_t stream) {
    const float* x   = (const float*)d_in[0];
    const float* w_l = (const float*)d_in[1];
    const float* s_l = (const float*)d_in[2];
    const float* o_l = (const float*)d_in[3];
    const float* w_r = (const float*)d_in[4];
    const float* s_r = (const float*)d_in[5];
    const float* o_r = (const float*)d_in[6];
    const float* w_t = (const float*)d_in[7];
    const float* s_t = (const float*)d_in[8];
    const float* o_t = (const float*)d_in[9];
    const float* w_b = (const float*)d_in[10];
    const float* s_b = (const float*)d_in[11];
    const float* o_b = (const float*)d_in[12];
    const float* w2  = (const float*)d_in[13];
    const float* s2  = (const float*)d_in[14];
    const float* o2  = (const float*)d_in[15];
    const float* w1  = (const float*)d_in[16];
    const float* s1  = (const float*)d_in[17];
    const float* o1  = (const float*)d_in[18];

    // ---- workspace layout (f16) ----
    _Float16* xp    = (_Float16*)d_ws;                                  // 8*130*136*256  = 72.4 MB
    _Float16* y16   = xp  + (size_t)BATCH * HP * WP * CIN;              // 8*512*128*128  = 134 MB
    _Float16* sp    = y16 + (size_t)BATCH * 4 * CMID * HW * HW;         // 8*130*136*128  = 36.2 MB
    _Float16* wp1   = sp  + (size_t)BATCH * HP * WP * CMID;             // 1,179,648
    _Float16* wp2   = wp1 + (size_t)NW1;                                //   294,912
    _Float16* wp1x1 = wp2 + (size_t)NW2;                                //    65,536

    pack_x_kernel<<<dim3(HP, BATCH), 256, 0, stream>>>(x, xp);

    pack_w_kernel<<<(NW1 + NW2 + NW3 + 255) / 256, 256, 0, stream>>>(
        w_l, s_l, w_r, s_r, w_t, s_t, w_b, s_b, w2, s2, w1, s1, wp1, wp2, wp1x1);

    branch_conv_kernel<<<dim3(8, HW, BATCH), 256, 0, stream>>>(
        xp, wp1, o_l, o_r, o_t, o_b, y16);

    corner_pool_kernel<<<dim3(CMID, BATCH), 128, 0, stream>>>(y16, sp);

    fuse_conv_kernel<<<dim3(4, HW, BATCH), 256, 0, stream>>>(
        sp, xp, wp2, wp1x1, o2, o1, (float*)d_out);
}
